// GCN_hw_47502338294129
// MI455X (gfx1250) — matrix operand order repaired and checked by reading, not compiled, not run on hardware
//
#include <hip/hip_runtime.h>
#include <hip/hip_bf16.h>
#include <stdint.h>

// ---------------------------------------------------------------------------
// Problem constants (from reference)
// ---------------------------------------------------------------------------
#define N_NODES 10000   // rows of A, K of GEMM1
#define HIDC    512     // hidden dim
#define BATCH   4096    // node pairs
#define KPAD    10048   // N_NODES padded to multiple of 64 (zero-filled)

// ---------------------------------------------------------------------------
// Workspace layout (bytes). Total ~49.6 MB
// ---------------------------------------------------------------------------
#define OFF_W1T   ((size_t)0)                    // W1^T bf16 [512 x KPAD] = 10,289,152
#define OFF_WHB0  (OFF_W1T  + (size_t)10289152)  // Wh0 bf16 [512 x 512]   =    524,288
#define OFF_WGB0  (OFF_WHB0 + (size_t)524288)
#define OFF_WHB1  (OFF_WGB0 + (size_t)524288)
#define OFF_WGB1  (OFF_WHB1 + (size_t)524288)
#define OFF_L1    (OFF_WGB1 + (size_t)524288)    // layer1 f32 [10000 x 512] = 20,480,000
#define OFF_XA    (OFF_L1   + (size_t)20480000)  // x ping  f32 [4096 x 512] =  8,388,608
#define OFF_XB    (OFF_XA   + (size_t)8388608)   // x pong  f32 [4096 x 512] =  8,388,608

// ---------------------------------------------------------------------------
// CDNA5 WMMA types (wave32; 16x16x32 bf16 -> f32)
// ---------------------------------------------------------------------------
typedef __attribute__((ext_vector_type(16))) __bf16 v16bf;
typedef __attribute__((ext_vector_type(8)))  float  v8f;

union Frag16 { unsigned int u[8]; v16bf v; };

__device__ __forceinline__ unsigned short f2bf(float f) {
  unsigned int u = __float_as_uint(f);
  u += 0x7FFFu + ((u >> 16) & 1u);   // round-to-nearest-even
  return (unsigned short)(u >> 16);
}
__device__ __forceinline__ unsigned int pack2(float lo, float hi) {
  return (unsigned int)f2bf(lo) | ((unsigned int)f2bf(hi) << 16);
}

// ---------------------------------------------------------------------------
// Optional CDNA5 async global->LDS staging (ASYNCcnt path).
// Builtin signature (from hipcc diagnostic): (v4i as1*, v4i as3*, Ii, Ii)
// ---------------------------------------------------------------------------
#if defined(__has_builtin)
#if __has_builtin(__builtin_amdgcn_global_load_async_to_lds_b128)
#define HAVE_ASYNC_LDS 1
#endif
#endif

typedef int v4i_vs __attribute__((vector_size(16)));
typedef __attribute__((address_space(1))) v4i_vs as1_v4i;
typedef __attribute__((address_space(3))) v4i_vs as3_v4i;

__device__ __forceinline__ as1_v4i* to_glob_v4(const void* p) {
  return (as1_v4i*)(unsigned long long)p;
}
__device__ __forceinline__ as3_v4i* to_lds_v4(void* p) {
  // generic LDS pointer: low 32 bits are the LDS byte offset (ISA aperture rule)
  return (as3_v4i*)(unsigned int)(unsigned long long)p;
}

__device__ __forceinline__ void wait_async_zero() {
#if defined(__has_builtin)
#if __has_builtin(__builtin_amdgcn_s_wait_asynccnt)
  __builtin_amdgcn_s_wait_asynccnt(0);
#else
  asm volatile("s_wait_asynccnt 0x0" ::: "memory");
#endif
#else
  asm volatile("s_wait_asynccnt 0x0" ::: "memory");
#endif
}

// LDS strides (u16 units). 40 shorts = 80B rows: 16B-aligned fragment reads
// (-> ds_load_b128) and 20-bank row offsets (conflict-free across 16 lanes).
#define LDA2 40
#define LDBK 40
#define LDB  36   // highway kernel strides (round-1 layout, kept)
#define LDAH 36

// ---------------------------------------------------------------------------
// f32 -> bf16 conversion (vectorized; n divisible by 4 in all uses)
// ---------------------------------------------------------------------------
__global__ __launch_bounds__(256)
void cvt_bf16_kernel(const float* __restrict__ src, unsigned short* __restrict__ dst, int n) {
  const int i = (blockIdx.x * 256 + threadIdx.x) * 4;
  if (i + 4 <= n) {
    const float4 f = *reinterpret_cast<const float4*>(src + i);
    uint2 p;
    p.x = pack2(f.x, f.y);
    p.y = pack2(f.z, f.w);
    *reinterpret_cast<uint2*>(dst + i) = p;
  } else {
    for (int j = i; j < n; ++j) dst[j] = f2bf(src[j]);
  }
}

// ---------------------------------------------------------------------------
// W1 [10000 x 512] f32  ->  W1T [512 x KPAD] bf16 (transposed, zero K-pad).
// LDS-tiled 64x64 transpose: coalesced f32 reads, coalesced bf16 writes.
// ---------------------------------------------------------------------------
__global__ __launch_bounds__(256)
void cvt_transpose_w1_kernel(const float* __restrict__ W1, unsigned short* __restrict__ W1T) {
  __shared__ unsigned short T[64][72];   // [n][k] tile; 144B rows keep uint4 reads aligned
  const int t  = threadIdx.x;
  const int k0 = blockIdx.x * 64;
  const int n0 = blockIdx.y * 64;

  // phase 1: read 64(k) x 64(n) f32, store transposed bf16 into LDS
  const int rk = t >> 2;          // 0..63
  const int rn = (t & 3) * 16;    // 0,16,32,48
  const int gk = k0 + rk;
  #pragma unroll
  for (int i = 0; i < 4; ++i) {
    float4 f = make_float4(0.f, 0.f, 0.f, 0.f);
    if (gk < N_NODES)
      f = *reinterpret_cast<const float4*>(W1 + (size_t)gk * HIDC + n0 + rn + i * 4);
    T[rn + i * 4 + 0][rk] = f2bf(f.x);
    T[rn + i * 4 + 1][rk] = f2bf(f.y);
    T[rn + i * 4 + 2][rk] = f2bf(f.z);
    T[rn + i * 4 + 3][rk] = f2bf(f.w);
  }
  __syncthreads();

  // phase 2: write W1T rows (coalesced 32B per thread)
  const int wn = t >> 2;
  const int wk = (t & 3) * 16;
  const uint4 q0 = *reinterpret_cast<const uint4*>(&T[wn][wk]);
  const uint4 q1 = *reinterpret_cast<const uint4*>(&T[wn][wk + 8]);
  unsigned short* dst = W1T + (size_t)(n0 + wn) * KPAD + k0 + wk;
  *reinterpret_cast<uint4*>(dst + 0) = q0;
  *reinterpret_cast<uint4*>(dst + 8) = q1;
}

// ---------------------------------------------------------------------------
// GEMM1: layer1 = relu(A_f32 @ W1), M=10000, N=512, K=10000.
// Workgroup: 256 threads = 8 waves (2 M-strips x 4 N-strips);
// tile 32(M) x 256(N); K-step 32.  A re-read only 2x from HBM (~800 MB total);
// W1T stays L2-resident.  A converted f32->bf16 during LDS staging;
// B rows copied straight from W1T (async-to-LDS when available).
// ---------------------------------------------------------------------------
__global__ __launch_bounds__(256)
void gemm1_relu_kernel(const float* __restrict__ A,
                       const unsigned short* __restrict__ W1T,  // [512][KPAD] bf16
                       float* __restrict__ Cout)                // [10000 x 512]
{
  __shared__ unsigned short As[32 * LDA2];     //  2,560 B
  __shared__ unsigned short Bs[256 * LDBK];    // 20,480 B

  const int t     = threadIdx.x;
  const int lane  = t & 31;
  const int wave  = t >> 5;
  const int waveM = wave >> 2;   // 0..1
  const int waveN = wave & 3;    // 0..3
  const int m0 = blockIdx.x * 32;
  const int n0 = blockIdx.y * 256;

  v8f acc[4];
  #pragma unroll
  for (int f = 0; f < 4; ++f)
    acc[f] = (v8f){0.f,0.f,0.f,0.f,0.f,0.f,0.f,0.f};

  // A staging: 32x32 f32 tile, one float4 per thread
  const int arow = t >> 3;          // 0..31
  const int acol = (t & 7) * 4;     // 0..28
  const int gm   = m0 + arow;
  const float* asrc0 = A + (size_t)gm * N_NODES;

  for (int k0 = 0; k0 < N_NODES; k0 += 32) {
    // ---- stage A tile (f32 -> bf16) ----
    {
      const float* asrc = asrc0 + k0 + acol;
      uint2 p;
      if (gm < N_NODES && (k0 + acol + 4) <= N_NODES) {
        __builtin_prefetch(asrc + 32, 0, 1);   // next K tile -> global_prefetch_b8
        const float4 f = *reinterpret_cast<const float4*>(asrc);
        p.x = pack2(f.x, f.y);
        p.y = pack2(f.z, f.w);
      } else {
        float x0 = 0.f, x1 = 0.f, x2 = 0.f, x3 = 0.f;
        if (gm < N_NODES) {
          const int kk = k0 + acol;
          if (kk     < N_NODES) x0 = asrc[0];
          if (kk + 1 < N_NODES) x1 = asrc[1];
          if (kk + 2 < N_NODES) x2 = asrc[2];
          if (kk + 3 < N_NODES) x3 = asrc[3];
        }
        p.x = pack2(x0, x1);
        p.y = pack2(x2, x3);
      }
      *reinterpret_cast<uint2*>(&As[arow * LDA2 + acol]) = p;
    }
    // ---- stage B tile: Bs[n][k] row-copy from W1T (no transpose, no guards) ----
    {
      const unsigned short* gsrc = W1T + (size_t)(n0 + t) * KPAD + k0;
      unsigned short* ldst = &Bs[t * LDBK];
#if defined(HAVE_ASYNC_LDS)
      #pragma unroll
      for (int i = 0; i < 4; ++i)
        __builtin_amdgcn_global_load_async_to_lds_b128(
            to_glob_v4(gsrc + i * 8), to_lds_v4(ldst + i * 8), 0, 0);
      wait_async_zero();
#else
      uint4 q[4];
      #pragma unroll
      for (int i = 0; i < 4; ++i)
        q[i] = *reinterpret_cast<const uint4*>(gsrc + i * 8);
      #pragma unroll
      for (int i = 0; i < 4; ++i)
        *reinterpret_cast<uint4*>(ldst + i * 8) = q[i];
#endif
    }
    __syncthreads();

    // ---- A fragment (16x32 bf16, k = 16*half + s; 2x 16B-aligned ds reads) ----
    Frag16 af;
    {
      const int lm = lane & 15, lh = lane >> 4;
      const unsigned short* ar = &As[(waveM * 16 + lm) * LDA2];
      #pragma unroll
      for (int v = 0; v < 8; ++v) {
        const int kv = lh * 16 + 2 * v;
        af.u[v] = *reinterpret_cast<const unsigned int*>(ar + kv);
      }
    }
    // ---- 4 B fragments (32x16 bf16) + WMMA ----
    #pragma unroll
    for (int f = 0; f < 4; ++f) {
      const int ln = lane & 15, lh = lane >> 4;
      Frag16 bf;
      const unsigned short* br = &Bs[(waveN * 64 + f * 16 + ln) * LDBK + lh * 16];
      #pragma unroll
      for (int v = 0; v < 8; ++v)
        bf.u[v] = *reinterpret_cast<const unsigned int*>(br + 2 * v);
      acc[f] = __builtin_amdgcn_wmma_f32_16x16x32_bf16(
                   false, af.v, false, bf.v, (short)0, acc[f], false, false);
    }
    __syncthreads();
  }

  // ---- epilogue: relu + store (C layout: M = r + 8*half, N = lane&15) ----
  const int lm = lane & 15, lh = lane >> 4;
  #pragma unroll
  for (int f = 0; f < 4; ++f) {
    const int col = n0 + waveN * 64 + f * 16 + lm;
    #pragma unroll
    for (int r = 0; r < 8; ++r) {
      const int row = m0 + waveM * 16 + lh * 8 + r;
      if (row < N_NODES)
        Cout[(size_t)row * HIDC + col] = fmaxf(acc[f][r], 0.0f);
    }
  }
}

// ---------------------------------------------------------------------------
// Pair features: x0[b][j] = (layer1[ix[b]][j] - layer1[iy[b]][j])^2 / sqrt(512)
// ---------------------------------------------------------------------------
__global__ __launch_bounds__(256)
void pair_feat_kernel(const float* __restrict__ L1,
                      const int* __restrict__ ix, const int* __restrict__ iy,
                      float* __restrict__ X)
{
  const int gid = blockIdx.x * 256 + threadIdx.x;   // over 4096*128 float4s
  const int b = gid >> 7;
  const int j = (gid & 127) << 2;
  const float4 ex = *reinterpret_cast<const float4*>(L1 + (size_t)ix[b] * HIDC + j);
  const float4 ey = *reinterpret_cast<const float4*>(L1 + (size_t)iy[b] * HIDC + j);
  const float s = 0.04419417382415922f;             // 1/sqrt(512)
  float4 o;
  float d;
  d = ex.x - ey.x; o.x = d * d * s;
  d = ex.y - ey.y; o.y = d * d * s;
  d = ex.z - ey.z; o.z = d * d * s;
  d = ex.w - ey.w; o.w = d * d * s;
  *reinterpret_cast<float4*>(X + (size_t)b * HIDC + j) = o;
}

// ---------------------------------------------------------------------------
// Highway layer (fused dual-GEMM): h = relu(x@Wh + bh); t = sigmoid(x@Wg + bg);
// out = h*t + x*(1-t).  M=4096, N=512, K=512; tile 128x64, 8 waves x 16M.
// ---------------------------------------------------------------------------
__global__ __launch_bounds__(256)
void highway_kernel(const float* __restrict__ Xin,
                    const unsigned short* __restrict__ Whb,
                    const float* __restrict__ bh,
                    const unsigned short* __restrict__ Wgb,
                    const float* __restrict__ bg,
                    float* __restrict__ Xout)
{
  __shared__ unsigned short As [128 * LDAH];
  __shared__ unsigned short Bhs[64 * LDB];
  __shared__ unsigned short Bgs[64 * LDB];

  const int t    = threadIdx.x;
  const int lane = t & 31;
  const int wave = t >> 5;
  const int m0 = blockIdx.x * 128;
  const int n0 = blockIdx.y * 64;

  v8f acch[4], acct[4];
  #pragma unroll
  for (int f = 0; f < 4; ++f) {
    acch[f] = (v8f){0.f,0.f,0.f,0.f,0.f,0.f,0.f,0.f};
    acct[f] = (v8f){0.f,0.f,0.f,0.f,0.f,0.f,0.f,0.f};
  }

  const int arow = t >> 1;
  const int acol = (t & 1) * 16;
  const float* asrc0 = Xin + (size_t)(m0 + arow) * HIDC;
  const int bk = t >> 3;
  const int bn = (t & 7) * 8;

  for (int k0 = 0; k0 < HIDC; k0 += 32) {    // 16 iterations, no tails
    {
      unsigned int* adst = reinterpret_cast<unsigned int*>(&As[arow * LDAH + acol]);
      const float* asrc  = asrc0 + k0 + acol;
      if (k0 + 32 < HIDC) __builtin_prefetch(asrc + 32, 0, 1);
      #pragma unroll
      for (int i = 0; i < 4; ++i) {
        const float4 f = *reinterpret_cast<const float4*>(asrc + i * 4);
        adst[i * 2 + 0] = pack2(f.x, f.y);
        adst[i * 2 + 1] = pack2(f.z, f.w);
      }
    }
    {
      const size_t boff = (size_t)(k0 + bk) * HIDC + n0 + bn;
      const uint4 qh = *reinterpret_cast<const uint4*>(Whb + boff);
      const uint4 qg = *reinterpret_cast<const uint4*>(Wgb + boff);
      const unsigned int rh[4] = {qh.x, qh.y, qh.z, qh.w};
      const unsigned int rg[4] = {qg.x, qg.y, qg.z, qg.w};
      #pragma unroll
      for (int j = 0; j < 4; ++j) {
        Bhs[(bn + 2 * j    ) * LDB + bk] = (unsigned short)(rh[j] & 0xFFFFu);
        Bhs[(bn + 2 * j + 1) * LDB + bk] = (unsigned short)(rh[j] >> 16);
        Bgs[(bn + 2 * j    ) * LDB + bk] = (unsigned short)(rg[j] & 0xFFFFu);
        Bgs[(bn + 2 * j + 1) * LDB + bk] = (unsigned short)(rg[j] >> 16);
      }
    }
    __syncthreads();

    Frag16 af;
    {
      const int lm = lane & 15, lh = lane >> 4;
      const unsigned short* ar = &As[(wave * 16 + lm) * LDAH];
      #pragma unroll
      for (int v = 0; v < 8; ++v) {
        const int kv = lh * 16 + 2 * v;
        af.u[v] = *reinterpret_cast<const unsigned int*>(ar + kv);
      }
    }
    #pragma unroll
    for (int f = 0; f < 4; ++f) {
      const int ln = lane & 15, lh = lane >> 4;
      Frag16 bfh, bfg;
      const unsigned short* brh = &Bhs[(f * 16 + ln) * LDB + lh * 16];
      const unsigned short* brg = &Bgs[(f * 16 + ln) * LDB + lh * 16];
      #pragma unroll
      for (int v = 0; v < 8; ++v) {
        bfh.u[v] = *reinterpret_cast<const unsigned int*>(brh + 2 * v);
        bfg.u[v] = *reinterpret_cast<const unsigned int*>(brg + 2 * v);
      }
      acch[f] = __builtin_amdgcn_wmma_f32_16x16x32_bf16(
                    false, af.v, false, bfh.v, (short)0, acch[f], false, false);
      acct[f] = __builtin_amdgcn_wmma_f32_16x16x32_bf16(
                    false, af.v, false, bfg.v, (short)0, acct[f], false, false);
    }
    __syncthreads();
  }

  // ---- epilogue: gate + residual ----
  const int lm = lane & 15, lh = lane >> 4;
  #pragma unroll
  for (int f = 0; f < 4; ++f) {
    const int col = n0 + f * 16 + lm;
    const float bhv = bh[col];
    const float bgv = bg[col];
    #pragma unroll
    for (int r = 0; r < 8; ++r) {
      const int row = m0 + wave * 16 + lh * 8 + r;
      const float h  = fmaxf(acch[f][r] + bhv, 0.0f);
      const float z  = acct[f][r] + bgv;
      const float tt = 1.0f / (1.0f + __expf(-z));
      const float xi = Xin[(size_t)row * HIDC + col];
      Xout[(size_t)row * HIDC + col] = h * tt + xi * (1.0f - tt);
    }
  }
}

// ---------------------------------------------------------------------------
// Final projection (512 -> 2) + log_softmax; one wave per row (wave32).
// ---------------------------------------------------------------------------
__global__ __launch_bounds__(256)
void logits_kernel(const float* __restrict__ X,
                   const float* __restrict__ Wl,
                   const float* __restrict__ bl,
                   float* __restrict__ out)
{
  const int wave = threadIdx.x >> 5;
  const int lane = threadIdx.x & 31;
  const int b = blockIdx.x * 8 + wave;
  float s0 = 0.0f, s1 = 0.0f;
  #pragma unroll 4
  for (int j = lane; j < HIDC; j += 32) {
    const float v = X[(size_t)b * HIDC + j];
    s0 += v * Wl[j * 2 + 0];
    s1 += v * Wl[j * 2 + 1];
  }
  #pragma unroll
  for (int off = 16; off > 0; off >>= 1) {
    s0 += __shfl_xor(s0, off, 32);
    s1 += __shfl_xor(s1, off, 32);
  }
  if (lane == 0) {
    const float l0 = s0 + bl[0];
    const float l1 = s1 + bl[1];
    const float m  = fmaxf(l0, l1);
    const float lse = m + __logf(__expf(l0 - m) + __expf(l1 - m));
    out[b * 2 + 0] = l0 - lse;
    out[b * 2 + 1] = l1 - lse;
  }
}

// ---------------------------------------------------------------------------
// Launch
// ---------------------------------------------------------------------------
extern "C" void kernel_launch(void* const* d_in, const int* in_sizes, int n_in,
                              void* d_out, int out_size, void* d_ws, size_t ws_size,
                              hipStream_t stream) {
  const float* A   = (const float*)d_in[0];
  const float* W1  = (const float*)d_in[1];
  const float* Wh0 = (const float*)d_in[2];
  const float* bh0 = (const float*)d_in[3];
  const float* Wg0 = (const float*)d_in[4];
  const float* bg0 = (const float*)d_in[5];
  const float* Wh1 = (const float*)d_in[6];
  const float* bh1 = (const float*)d_in[7];
  const float* Wg1 = (const float*)d_in[8];
  const float* bg1 = (const float*)d_in[9];
  const float* Wl  = (const float*)d_in[10];
  const float* bl  = (const float*)d_in[11];
  const int*   ix  = (const int*)d_in[12];
  const int*   iy  = (const int*)d_in[13];

  char* ws = (char*)d_ws;
  unsigned short* W1T  = (unsigned short*)(ws + OFF_W1T);
  unsigned short* Whb0 = (unsigned short*)(ws + OFF_WHB0);
  unsigned short* Wgb0 = (unsigned short*)(ws + OFF_WGB0);
  unsigned short* Whb1 = (unsigned short*)(ws + OFF_WHB1);
  unsigned short* Wgb1 = (unsigned short*)(ws + OFF_WGB1);
  float* L1 = (float*)(ws + OFF_L1);
  float* xA = (float*)(ws + OFF_XA);
  float* xB = (float*)(ws + OFF_XB);

  // 1) weight conversions (W1 transposed + K-padded for the WMMA B operand)
  {
    dim3 gt(KPAD / 64, HIDC / 64);   // (157, 8)
    cvt_transpose_w1_kernel<<<gt, 256, 0, stream>>>(W1, W1T);
  }
  cvt_bf16_kernel<<<(HIDC * HIDC) / (256 * 4), 256, 0, stream>>>(Wh0, Whb0, HIDC * HIDC);
  cvt_bf16_kernel<<<(HIDC * HIDC) / (256 * 4), 256, 0, stream>>>(Wg0, Wgb0, HIDC * HIDC);
  cvt_bf16_kernel<<<(HIDC * HIDC) / (256 * 4), 256, 0, stream>>>(Wh1, Whb1, HIDC * HIDC);
  cvt_bf16_kernel<<<(HIDC * HIDC) / (256 * 4), 256, 0, stream>>>(Wg1, Wgb1, HIDC * HIDC);

  // 2) GCN layer: layer1 = relu(A @ W1)   (M tiles: ceil(10000/32)=313, N tiles: 2)
  {
    dim3 grid(313, 2);
    gemm1_relu_kernel<<<grid, 256, 0, stream>>>(A, W1T, L1);
  }

  // 3) pair features
  pair_feat_kernel<<<(BATCH * (HIDC / 4)) / 256, 256, 0, stream>>>(L1, ix, iy, xA);

  // 4) two highway layers (ping-pong xA -> xB -> xA)
  {
    dim3 grid(BATCH / 128, HIDC / 64);
    highway_kernel<<<grid, 256, 0, stream>>>(xA, Whb0, bh0, Wgb0, bg0, xB);
    highway_kernel<<<grid, 256, 0, stream>>>(xB, Whb1, bh1, Wgb1, bg1, xA);
  }

  // 5) logits + log_softmax -> d_out [4096 x 2]
  logits_kernel<<<BATCH / 8, 256, 0, stream>>>(xA, Wl, bl, (float*)d_out);
}
